// MILModel7_74801150427709
// MI455X (gfx1250) — matrix operand order repaired and checked by reading, not compiled, not run on hardware
//
#include <hip/hip_runtime.h>

// ---------------------------------------------------------------------------
// CDNA5 (gfx1250) fused MIL model:
//   scores = softmax_i( W . relu(V . ffeat) )   (pairwise term cancels exactly)
//   ffeat  = sigmoid(amo . tanh(amv.vo1 + ama.ax)) * ax + vo1
// All GEMMs run on v_wmma_f32_16x16x32_bf16 (f32 accumulate).
// f32->bf16 conversions use v_cvt_pk_bf16_f32 via inline asm (1 op / 2 elems).
// ---------------------------------------------------------------------------

typedef __attribute__((ext_vector_type(16))) __bf16         v16bf;
typedef __attribute__((ext_vector_type(16))) unsigned short v16u;
typedef __attribute__((ext_vector_type(8)))  unsigned int   v8u;
typedef __attribute__((ext_vector_type(8)))  float          v8f;

union FragU  { v16u u; v16bf b; };
union FragUD { v8u  d; v16bf b; };

// Hardware packed conversion: D[15:0]=bf16(lo), D[31:16]=bf16(hi), RNE.
__device__ __forceinline__ unsigned int cvt_pk_bf16(float lo, float hi) {
  unsigned int r;
  asm("v_cvt_pk_bf16_f32 %0, %1, %2" : "=v"(r) : "v"(lo), "v"(hi));
  return r;
}

__device__ __forceinline__ unsigned short f2bf(float f) {   // host-side-ish path (pack kernel)
  unsigned int u = __float_as_uint(f);
  unsigned int r = (u + 0x7FFFu + ((u >> 16) & 1u)) >> 16;  // round-nearest-even
  return (unsigned short)r;
}
__device__ __forceinline__ float bf2f(unsigned short h) {
  return __uint_as_float(((unsigned int)h) << 16);
}

// A-fragment (16x32 bf16, MxK) index for element (m,k) within one wave region:
//   lanes 0-15 : M=lane, K = {kt*32+0..7, kt*32+16..23}
//   lanes 16-31: M=lane-16, K = {kt*32+8..15, kt*32+24..31}
__device__ __forceinline__ int frag_idx(int m, int k) {
  int lane_t = (m & 15) | (((k >> 3) & 1) << 4);
  int slot   = (k & 7)  | (((k >> 4) & 1) << 3);
  return ((k >> 5) << 9) + lane_t * 16 + slot;
}

// ---------------------------------------------------------------------------
// Pack weight W (Nout x Kin, row-major f32, used as B = W^T) into bf16
// B-fragment layout: frag[kt][nt][lane][slot],
//   lane: N = nt*16 + (lane&15), K = kt*32 + 8*(lane>>4) + (slot&7) + 16*(slot>>3)
// ---------------------------------------------------------------------------
__global__ void pack_b_frags(const float* __restrict__ W,
                             unsigned short* __restrict__ out,
                             int Nout, int Kin) {
  int idx = blockIdx.x * blockDim.x + threadIdx.x;
  int total = Nout * Kin;
  if (idx >= total) return;
  int slot = idx & 15;
  int lane = (idx >> 4) & 31;
  int fr   = idx >> 9;
  int NT   = Nout >> 4;
  int nt   = fr % NT;
  int kt   = fr / NT;
  int n = nt * 16 + (lane & 15);
  int k = kt * 32 + ((lane >> 4) << 3) + (slot & 7) + ((slot >> 3) << 4);
  out[idx] = f2bf(W[n * Kin + k]);
}

// Direct global->register A-fragment: 16 tokens (rows) x 32 K of f32 -> bf16.
__device__ __forceinline__ v16bf load_a_global(const float* __restrict__ row,
                                               int kbase, int hi) {
  const float4* p0 = (const float4*)(row + kbase + hi * 8);
  const float4* p1 = (const float4*)(row + kbase + 16 + hi * 8);
  float4 a0 = p0[0], a1 = p0[1];
  float4 b0 = p1[0], b1 = p1[1];
  FragUD f;
  f.d[0] = cvt_pk_bf16(a0.x, a0.y);
  f.d[1] = cvt_pk_bf16(a0.z, a0.w);
  f.d[2] = cvt_pk_bf16(a1.x, a1.y);
  f.d[3] = cvt_pk_bf16(a1.z, a1.w);
  f.d[4] = cvt_pk_bf16(b0.x, b0.y);
  f.d[5] = cvt_pk_bf16(b0.z, b0.w);
  f.d[6] = cvt_pk_bf16(b1.x, b1.y);
  f.d[7] = cvt_pk_bf16(b1.z, b1.w);
  return f.b;
}

__device__ __forceinline__ v16bf load_frag_lds(const unsigned short* base,
                                               int kt, int lane) {
  FragU f;
  f.u = *(const v16u*)(base + (kt << 9) + lane * 16);
  return f.b;
}

__device__ __forceinline__ v16bf load_frag_glb(const unsigned short* base,
                                               int fridx, int lane) {
  FragU f;
  f.u = *(const v16u*)(base + (fridx << 9) + lane * 16);
  return f.b;
}

#define WMMA_BF16(A, B, C) \
  __builtin_amdgcn_wmma_f32_16x16x32_bf16(false, (A), false, (B), (short)0, (C), false, false)

// ---------------------------------------------------------------------------
// Main fused kernel: 256 thr = 8 waves; each wave owns 16 tokens, all columns.
// No cross-wave barriers needed (per-wave LDS regions only).
// ---------------------------------------------------------------------------
__global__ __launch_bounds__(256) void mil_main(
    const float* __restrict__ vfeat,   // (65536, 512)
    const float* __restrict__ afeat,   // (65536, 128)
    const float* __restrict__ vfc_b, const float* __restrict__ afc_b,
    const float* __restrict__ amv_b, const float* __restrict__ ama_b,
    const float* __restrict__ amo_w, const float* __restrict__ amo_b,
    const float* __restrict__ V_b,   const float* __restrict__ W_w,
    const unsigned short* __restrict__ wvfc,   // [16][8][512] bf16 frags
    const unsigned short* __restrict__ wafc,   // [4][8][512]
    const unsigned short* __restrict__ wamv,   // [4][8][512]
    const unsigned short* __restrict__ wama,   // [4][8][512]
    const unsigned short* __restrict__ wV,     // [4][4][512]
    float* __restrict__ ff_out,                // (65536, 128) f32
    float* __restrict__ s_out)                 // (65536)
{
  // Per-wave bf16 A-fragment stores: vo1 (later aliased as ffeat) and ax.
  __shared__ __align__(32) unsigned short s_vo1[8 * 2048];  // 32 KB (also ffeat)
  __shared__ __align__(32) unsigned short s_ax [8 * 2048];  // 32 KB

  const int tid  = threadIdx.x;
  const int wave = tid >> 5;
  const int lane = tid & 31;
  const int hi   = lane >> 4;     // half-wave selector
  const int j    = lane & 15;     // column-within-tile / row-within-tile index
  const int tok0 = blockIdx.x * 128 + wave * 16;

  unsigned short* myvo1 = s_vo1 + wave * 2048;
  unsigned short* myax  = s_ax  + wave * 2048;

  v8f zero;
  #pragma unroll
  for (int i = 0; i < 8; i++) zero[i] = 0.f;

  // ---------------- GEMM 1: vo1 = relu(vfeat @ vfc_w^T + vfc_b), K=512 -----
  v8f acc[8];
  #pragma unroll
  for (int nt = 0; nt < 8; nt++) acc[nt] = zero;
  {
    const float* vrow = vfeat + (size_t)(tok0 + j) * 512;
    #pragma unroll 2
    for (int kt = 0; kt < 16; kt++) {
      v16bf a = load_a_global(vrow, kt * 32, hi);
      #pragma unroll
      for (int nt = 0; nt < 8; nt++)
        acc[nt] = WMMA_BF16(a, load_frag_glb(wvfc, kt * 8 + nt, lane), acc[nt]);
    }
  }
  #pragma unroll
  for (int nt = 0; nt < 8; nt++) {
    float bias = vfc_b[nt * 16 + j];
    int k = nt * 16 + j;
    #pragma unroll
    for (int r = 0; r < 8; r += 2) {
      float v0 = fmaxf(acc[nt][r] + bias, 0.f);
      float v1 = fmaxf(acc[nt][r + 1] + bias, 0.f);
      unsigned int p = cvt_pk_bf16(v0, v1);
      myvo1[frag_idx(r     + 8 * hi, k)] = (unsigned short)p;
      myvo1[frag_idx(r + 1 + 8 * hi, k)] = (unsigned short)(p >> 16);
    }
  }

  // ---------------- GEMM 2: ax = relu(afeat @ afc_w^T + afc_b), K=128 ------
  v8f acc2[8];
  #pragma unroll
  for (int nt = 0; nt < 8; nt++) acc2[nt] = zero;
  {
    const float* arow = afeat + (size_t)(tok0 + j) * 128;
    #pragma unroll
    for (int kt = 0; kt < 4; kt++) {
      v16bf a = load_a_global(arow, kt * 32, hi);
      #pragma unroll
      for (int nt = 0; nt < 8; nt++)
        acc2[nt] = WMMA_BF16(a, load_frag_glb(wafc, kt * 8 + nt, lane), acc2[nt]);
    }
  }
  #pragma unroll
  for (int nt = 0; nt < 8; nt++) {
    float bias = afc_b[nt * 16 + j];
    int k = nt * 16 + j;
    #pragma unroll
    for (int r = 0; r < 8; r += 2) {
      float v0 = fmaxf(acc2[nt][r] + bias, 0.f);
      float v1 = fmaxf(acc2[nt][r + 1] + bias, 0.f);
      acc2[nt][r]     = v0;                              // keep f32 ax in regs
      acc2[nt][r + 1] = v1;
      unsigned int p = cvt_pk_bf16(v0, v1);
      myax[frag_idx(r     + 8 * hi, k)] = (unsigned short)p;
      myax[frag_idx(r + 1 + 8 * hi, k)] = (unsigned short)(p >> 16);
    }
  }

  // ---------------- GEMM 3: h = tanh(vo1@amv^T + ax@ama^T + biases) --------
  #pragma unroll
  for (int nt = 0; nt < 8; nt++) acc[nt] = zero;
  #pragma unroll
  for (int kt = 0; kt < 4; kt++) {
    v16bf av = load_frag_lds(myvo1, kt, lane);
    #pragma unroll
    for (int nt = 0; nt < 8; nt++)
      acc[nt] = WMMA_BF16(av, load_frag_glb(wamv, kt * 8 + nt, lane), acc[nt]);
    v16bf aa = load_frag_lds(myax, kt, lane);
    #pragma unroll
    for (int nt = 0; nt < 8; nt++)
      acc[nt] = WMMA_BF16(aa, load_frag_glb(wama, kt * 8 + nt, lane), acc[nt]);
  }

  // -------- att = sigmoid(amo_w . h + amo_b): 16-lane butterfly reduce -----
  float attv[8];
  {
    float wo[8], hb[8];
    #pragma unroll
    for (int nt = 0; nt < 8; nt++) {
      wo[nt] = amo_w[nt * 16 + j];
      hb[nt] = amv_b[nt * 16 + j] + ama_b[nt * 16 + j];
    }
    float ob = amo_b[0];
    #pragma unroll
    for (int r = 0; r < 8; r++) {
      float p = 0.f;
      #pragma unroll
      for (int nt = 0; nt < 8; nt++)
        p += tanhf(acc[nt][r] + hb[nt]) * wo[nt];
      p += __shfl_xor(p, 1, 32);
      p += __shfl_xor(p, 2, 32);
      p += __shfl_xor(p, 4, 32);
      p += __shfl_xor(p, 8, 32);
      attv[r] = 1.f / (1.f + __expf(-(p + ob)));
    }
  }

  // -------- ffeat = att*ax + vo1 : store f32 to ws, bf16 frag over vo1 -----
  // (read-before-write per element; myvo1 region becomes the ffeat frags)
  unsigned short* myff = myvo1;
  #pragma unroll
  for (int nt = 0; nt < 8; nt++) {
    int k = nt * 16 + j;
    #pragma unroll
    for (int r = 0; r < 8; r += 2) {
      int m0 = r + 8 * hi, m1 = r + 1 + 8 * hi;
      int fi0 = frag_idx(m0, k), fi1 = frag_idx(m1, k);
      float fv0 = attv[r]     * acc2[nt][r]     + bf2f(myvo1[fi0]);
      float fv1 = attv[r + 1] * acc2[nt][r + 1] + bf2f(myvo1[fi1]);
      unsigned int p = cvt_pk_bf16(fv0, fv1);
      myff[fi0] = (unsigned short)p;
      myff[fi1] = (unsigned short)(p >> 16);
      ff_out[(size_t)(tok0 + m0) * 128 + k] = fv0;
      ff_out[(size_t)(tok0 + m1) * 128 + k] = fv1;
    }
  }

  // ---------------- GEMM 4: f_v = relu(ffeat @ V_w^T + V_b), N=64 ----------
  #pragma unroll
  for (int nt = 0; nt < 4; nt++) acc[nt] = zero;
  #pragma unroll
  for (int kt = 0; kt < 4; kt++) {
    v16bf a = load_frag_lds(myff, kt, lane);
    #pragma unroll
    for (int nt = 0; nt < 4; nt++)
      acc[nt] = WMMA_BF16(a, load_frag_glb(wV, kt * 4 + nt, lane), acc[nt]);
  }

  // -------- s = W_w . f_v  (W_b cancels in the softmax; f_u == f_v) --------
  {
    float ww[4], vb[4];
    #pragma unroll
    for (int nt = 0; nt < 4; nt++) {
      ww[nt] = W_w[nt * 16 + j];
      vb[nt] = V_b[nt * 16 + j];
    }
    #pragma unroll
    for (int r = 0; r < 8; r++) {
      float p = 0.f;
      #pragma unroll
      for (int nt = 0; nt < 4; nt++)
        p += fmaxf(acc[nt][r] + vb[nt], 0.f) * ww[nt];
      p += __shfl_xor(p, 1, 32);
      p += __shfl_xor(p, 2, 32);
      p += __shfl_xor(p, 4, 32);
      p += __shfl_xor(p, 8, 32);
      if (j == r) s_out[tok0 + r + 8 * hi] = p;  // one writer per row
    }
  }
}

// ---------------------------------------------------------------------------
// Per-batch finalize: scores = softmax(s), zfeat = scores . ffeat, logits.
// ---------------------------------------------------------------------------
__global__ __launch_bounds__(128) void mil_finalize(
    const float* __restrict__ sbuf,    // (32, 2048)
    const float* __restrict__ ffeat,   // (32, 2048, 128)
    const float* __restrict__ cls_w,   // (2, 128)
    const float* __restrict__ cls_b,   // (2)
    float* __restrict__ out)           // scores (32*2048) ++ logits (32*2)
{
  __shared__ float sc[2048];
  __shared__ float red[128];
  const int b = blockIdx.x, tid = threadIdx.x;
  const float* srow = sbuf + b * 2048;

  float mx = -3.4e38f;
  for (int t = tid; t < 2048; t += 128) { float v = srow[t]; sc[t] = v; mx = fmaxf(mx, v); }
  red[tid] = mx; __syncthreads();
  for (int s = 64; s > 0; s >>= 1) { if (tid < s) red[tid] = fmaxf(red[tid], red[tid + s]); __syncthreads(); }
  float MX = red[0]; __syncthreads();

  float sum = 0.f;
  for (int t = tid; t < 2048; t += 128) { float e = __expf(sc[t] - MX); sc[t] = e; sum += e; }
  red[tid] = sum; __syncthreads();
  for (int s = 64; s > 0; s >>= 1) { if (tid < s) red[tid] += red[tid + s]; __syncthreads(); }
  float inv = 1.f / red[0]; __syncthreads();

  for (int t = tid; t < 2048; t += 128) { float p = sc[t] * inv; sc[t] = p; out[b * 2048 + t] = p; }
  __syncthreads();

  // zfeat column tid, then logits
  const float* fb = ffeat + (size_t)b * 2048 * 128;
  float zf = 0.f;
  for (int t = 0; t < 2048; t++) zf = fmaf(sc[t], fb[(size_t)t * 128 + tid], zf);
  red[tid] = zf; __syncthreads();
  if (tid < 2) {
    float a = cls_b[tid];
    for (int k = 0; k < 128; k++) a = fmaf(red[k], cls_w[tid * 128 + k], a);
    out[32 * 2048 + b * 2 + tid] = a;
  }
}

// ---------------------------------------------------------------------------
extern "C" void kernel_launch(void* const* d_in, const int* in_sizes, int n_in,
                              void* d_out, int out_size, void* d_ws, size_t ws_size,
                              hipStream_t stream) {
  (void)in_sizes; (void)n_in; (void)out_size; (void)ws_size;
  const float* vfeat = (const float*)d_in[0];
  const float* afeat = (const float*)d_in[1];
  const float* vfc_w = (const float*)d_in[2];
  const float* vfc_b = (const float*)d_in[3];
  const float* afc_w = (const float*)d_in[4];
  const float* afc_b = (const float*)d_in[5];
  const float* amv_w = (const float*)d_in[6];
  const float* amv_b = (const float*)d_in[7];
  const float* ama_w = (const float*)d_in[8];
  const float* ama_b = (const float*)d_in[9];
  const float* amo_w = (const float*)d_in[10];
  const float* amo_b = (const float*)d_in[11];
  const float* V_w   = (const float*)d_in[12];
  const float* V_b   = (const float*)d_in[13];
  const float* W_w   = (const float*)d_in[14];
  // d_in[15] = W_b : cancels in softmax over i, unused.
  const float* cls_w = (const float*)d_in[16];
  const float* cls_b = (const float*)d_in[17];

  char* ws = (char*)d_ws;
  unsigned short* wvfc = (unsigned short*)(ws);            // 131072 B
  unsigned short* wafc = (unsigned short*)(ws + 131072);   //  32768 B
  unsigned short* wamv = (unsigned short*)(ws + 163840);   //  32768 B
  unsigned short* wama = (unsigned short*)(ws + 196608);   //  32768 B
  unsigned short* wV   = (unsigned short*)(ws + 229376);   //  16384 B
  float* ffbuf = (float*)(ws + 262144);                    // 33.55 MB
  float* sbuf  = (float*)(ws + 262144 + (size_t)65536 * 128 * 4);  // 256 KB

  pack_b_frags<<<(128 * 512 + 255) / 256, 256, 0, stream>>>(vfc_w, wvfc, 128, 512);
  pack_b_frags<<<(128 * 128 + 255) / 256, 256, 0, stream>>>(afc_w, wafc, 128, 128);
  pack_b_frags<<<(128 * 128 + 255) / 256, 256, 0, stream>>>(amv_w, wamv, 128, 128);
  pack_b_frags<<<(128 * 128 + 255) / 256, 256, 0, stream>>>(ama_w, wama, 128, 128);
  pack_b_frags<<<(64  * 128 + 255) / 256, 256, 0, stream>>>(V_w,   wV,   64,  128);

  mil_main<<<512, 256, 0, stream>>>(vfeat, afeat, vfc_b, afc_b, amv_b, ama_b,
                                    amo_w, amo_b, V_b, W_w,
                                    wvfc, wafc, wamv, wama, wV, ffbuf, sbuf);

  mil_finalize<<<32, 128, 0, stream>>>(sbuf, ffbuf, cls_w, cls_b, (float*)d_out);
}
